// ApproxLinear_50440095924646
// MI455X (gfx1250) — matrix operand order repaired and checked by reading, not compiled, not run on hardware
//
#include <hip/hip_runtime.h>
#include <hip/hip_bf16.h>
#include <stdint.h>

// ---------------------------------------------------------------------------
// IVF inner-product ANN search (FAISS IVF8192,Flat, METRIC_INNER_PRODUCT)
//   B=1024 queries, D=128, V=200000 db vectors, P=8192 centroids,
//   NPROBE=32, TOPK=32.
// Pipeline:
//   k0: f32 -> bf16 conversion of x / centroids / W
//   k1: coarse GEMM qc = x . centroids^T          (v_wmma_f32_16x16x32_bf16)
//   k2: per-query top-NPROBE of qc -> probe bitset (8192 bits / query)
//   k3: main GEMM x . W^T, 64-query register tile, fused candidate masking,
//       survivors appended to global per-query candidate lists
//   k4: exact top-32 per query over its candidate list -> I (int32), D (f32)
// ---------------------------------------------------------------------------

typedef __attribute__((ext_vector_type(16))) __bf16 v16bf;
typedef __attribute__((ext_vector_type(8)))  __bf16 v8bf;
typedef __attribute__((ext_vector_type(8)))  float  v8f;

#define B_Q      1024
#define V_DB     200000
#define D_DIM    128
#define P_C      8192
#define NPROBE   32
#define TOPK     32
#define NSPLIT   50
#define SUBT_PER_SPLIT 250        // (V_DB/16)/NSPLIT
#define BITW     (P_C / 32)       // 256 bitset words per query
#define QT_TILES 4                // 16-row A tiles per wave
#define QT_QUERIES (QT_TILES * 16)  // 64 queries per block
#define GCAP     2048             // per-query candidate capacity (mean ~781)
#define NEGF     (-1.0e30f)
#define NEGINF   (-3.0e38f)

// ---------------- f32 -> bf16 (round to nearest even), stored as u16 --------
__global__ __launch_bounds__(256)
void cvt_f32_bf16(const float* __restrict__ in, unsigned short* __restrict__ out, int n) {
    int i = blockIdx.x * blockDim.x + threadIdx.x;
    int stride = gridDim.x * blockDim.x;
    for (; i < n; i += stride) {
        unsigned u = __float_as_uint(in[i]);
        unsigned r = u + 0x7FFFu + ((u >> 16) & 1u);
        out[i] = (unsigned short)(r >> 16);
    }
}

__global__ __launch_bounds__(256)
void zero_counts(int* __restrict__ c, int n) {
    int i = blockIdx.x * blockDim.x + threadIdx.x;
    if (i < n) c[i] = 0;
}

// ---------------- WMMA fragment loaders (ISA 16-bit A 16x32 / B 32x16) ------
// A: lanes 0-15 hold M=lane, K = kb+0..15; lanes 16-31 hold
//    M=lane-16, K = kb+16..31.  One contiguous 32B load per lane.
__device__ inline v16bf load_a_frag(const __bf16* __restrict__ row, int kbase, int half) {
    return *(const v16bf*)(row + kbase + half * 16);
}
// B: lane holds column N=lane%16, K = kb + (lane>=16 ? 16..31 : 0..15):
//    one contiguous 32B load from the W/centroid row acting as column N.
__device__ inline v16bf load_b_frag(const __bf16* __restrict__ row, int kbase, int half) {
    return *(const v16bf*)(row + kbase + half * 16);
}

__device__ inline v8f wmma_bf16(v16bf a, v16bf b, v8f c) {
    return __builtin_amdgcn_wmma_f32_16x16x32_bf16(false, a, false, b, (short)0, c, false, false);
}

// ---------------- k1: coarse GEMM qc[B,P] ----------------------------------
// grid = (P/64, B/16), block = 128 (4 waves); wave w -> 16x16 tile at n=w*16.
__global__ __launch_bounds__(128)
void coarse_gemm(const __bf16* __restrict__ xb, const __bf16* __restrict__ cb,
                 float* __restrict__ qc) {
    int t = threadIdx.x, lane = t & 31, wave = t >> 5;
    int half = lane >> 4, mn = lane & 15;
    int qbase = blockIdx.y * 16;
    int cbase = blockIdx.x * 64 + wave * 16;
    const __bf16* xrow = xb + (size_t)(qbase + mn) * D_DIM;
    const __bf16* crow = cb + (size_t)(cbase + mn) * D_DIM;
    v8f acc = {};
#pragma unroll
    for (int k = 0; k < 4; ++k) {
        v16bf a = load_a_frag(xrow, k * 32, half);
        v16bf b = load_b_frag(crow, k * 32, half);
        acc = wmma_bf16(a, b, acc);
    }
#pragma unroll
    for (int r = 0; r < 8; ++r) {
        int m = r + half * 8;   // C/D layout: VGPR r, lanes16-31 -> M=r+8
        qc[(size_t)(qbase + m) * P_C + cbase + mn] = acc[r];
    }
}

// ---------------- k2: top-NPROBE per query -> probe bitset ------------------
__global__ __launch_bounds__(256)
void topk_probe(const float* __restrict__ qc, unsigned* __restrict__ bits) {
    __shared__ float    row[P_C];      // 32 KB (LDS is 320KB/WGP on CDNA5)
    __shared__ unsigned words[BITW];
    __shared__ float    rv[256];
    __shared__ int      ri[256];
    int q = blockIdx.x, t = threadIdx.x;
    for (int i = t; i < P_C; i += 256) row[i] = qc[(size_t)q * P_C + i];
    for (int i = t; i < BITW; i += 256) words[i] = 0u;
    __syncthreads();
    for (int it = 0; it < NPROBE; ++it) {
        float best = NEGINF; int bi = -1;
        for (int i = t; i < P_C; i += 256) {
            float v = row[i];
            if (v > best) { best = v; bi = i; }
        }
        rv[t] = best; ri[t] = bi;
        __syncthreads();
        for (int s = 128; s > 0; s >>= 1) {
            if (t < s && rv[t + s] > rv[t]) { rv[t] = rv[t + s]; ri[t] = ri[t + s]; }
            __syncthreads();
        }
        if (t == 0) {
            int sel = ri[0];
            row[sel] = NEGINF;
            words[sel >> 5] |= (1u << (sel & 31));
        }
        __syncthreads();
    }
    for (int i = t; i < BITW; i += 256) bits[(size_t)q * BITW + i] = words[i];
}

// ---------------- k3: masked GEMM, 64-query register tile -------------------
// grid = (B/64, NSPLIT), block = 128 (4 waves). Each wave holds A fragments
// for 64 queries (4 x 16-row tiles) in registers and sweeps V-subtiles
// s = wave, wave+4, ... : per subtile, 8 global_load_b128 feed 16 WMMAs.
// Survivors of the probe-bitset test are appended to global candidate lists.
__global__ __launch_bounds__(128)
void scan_kernel(const __bf16* __restrict__ xb, const __bf16* __restrict__ wb,
                 const unsigned* __restrict__ bits, const int* __restrict__ assign,
                 int* __restrict__ ccount, float* __restrict__ candS,
                 int* __restrict__ candI) {
    __shared__ unsigned lbits[QT_QUERIES * BITW];   // 64 KB probe bitsets

    int t = threadIdx.x, lane = t & 31, wave = t >> 5;
    int half = lane >> 4, mn = lane & 15;
    int qbase = blockIdx.x * QT_QUERIES;
    int split = blockIdx.y;

    for (int i = t; i < QT_QUERIES * BITW; i += 128)
        lbits[i] = bits[(size_t)qbase * BITW + i];
    __syncthreads();

    // A fragments: 64 queries x 128 dims, resident in VGPRs for the sweep.
    v16bf a[QT_TILES][4];
#pragma unroll
    for (int qt = 0; qt < QT_TILES; ++qt) {
        const __bf16* xrow = xb + (size_t)(qbase + qt * 16 + mn) * D_DIM;
#pragma unroll
        for (int k = 0; k < 4; ++k) a[qt][k] = load_a_frag(xrow, k * 32, half);
    }

    int vt0 = split * SUBT_PER_SPLIT;
    for (int s = wave; s < SUBT_PER_SPLIT; s += 4) {
        int vbase = (vt0 + s) * 16;
        const __bf16* wrow = wb + (size_t)(vbase + mn) * D_DIM;
        if (s + 4 < SUBT_PER_SPLIT)
            __builtin_prefetch(wrow + 64 * D_DIM, 0, 1);   // global_prefetch
        v8f acc[QT_TILES] = {};
#pragma unroll
        for (int k = 0; k < 4; ++k) {
            v16bf b = load_b_frag(wrow, k * 32, half);     // shared across qtiles
#pragma unroll
            for (int qt = 0; qt < QT_TILES; ++qt)
                acc[qt] = wmma_bf16(a[qt][k], b, acc[qt]);
        }
        int v   = vbase + mn;           // one db vector per lane
        int aid = assign[v];
        int w_  = aid >> 5;
        unsigned msk = 1u << (aid & 31);
#pragma unroll
        for (int qt = 0; qt < QT_TILES; ++qt) {
#pragma unroll
            for (int r = 0; r < 8; ++r) {
                int m = qt * 16 + r + half * 8;     // query-in-block
                if (lbits[m * BITW + w_] & msk) {
                    int bq = qbase + m;
                    int p  = atomicAdd(&ccount[bq], 1);
                    if (p < GCAP) {
                        candS[(size_t)bq * GCAP + p] = acc[qt][r];
                        candI[(size_t)bq * GCAP + p] = v;
                    }
                }
            }
        }
    }
}

// ---------------- k4: exact top-32 per query over its candidates ------------
__global__ __launch_bounds__(256)
void final_topk(const int* __restrict__ ccount, const float* __restrict__ candS,
                const int* __restrict__ candI, int* __restrict__ outI,
                float* __restrict__ outD) {
    __shared__ float s[GCAP];
    __shared__ int   si[GCAP];
    __shared__ float rv[256];
    __shared__ int   ri[256];
    int q = blockIdx.x, t = threadIdx.x;
    int n = ccount[q];
    if (n > GCAP) n = GCAP;
    for (int i = t; i < n; i += 256) {
        s[i]  = candS[(size_t)q * GCAP + i];
        si[i] = candI[(size_t)q * GCAP + i];
    }
    __syncthreads();
    for (int it = 0; it < TOPK; ++it) {
        float best = NEGINF; int bi = -1;
        for (int i = t; i < n; i += 256) {
            float v = s[i];
            if (v > best) { best = v; bi = i; }
        }
        rv[t] = best; ri[t] = bi;
        __syncthreads();
        for (int st = 128; st > 0; st >>= 1) {
            if (t < st && rv[t + st] > rv[t]) { rv[t] = rv[t + st]; ri[t] = ri[t + st]; }
            __syncthreads();
        }
        if (t == 0) {
            int sel = ri[0];
            outD[(size_t)q * TOPK + it] = (sel >= 0) ? rv[0] : NEGF;
            outI[(size_t)q * TOPK + it] = (sel >= 0) ? si[sel] : 0;
            if (sel >= 0) s[sel] = NEGINF;
        }
        __syncthreads();
    }
}

// ---------------------------------------------------------------------------
extern "C" void kernel_launch(void* const* d_in, const int* in_sizes, int n_in,
                              void* d_out, int out_size, void* d_ws, size_t ws_size,
                              hipStream_t stream) {
    const float* x      = (const float*)d_in[0];   // [B, D]
    const float* W      = (const float*)d_in[1];   // [V, D]
    const float* C      = (const float*)d_in[2];   // [P, D]
    const int*   assign = (const int*)d_in[3];     // [V]
    (void)in_sizes; (void)n_in; (void)out_size; (void)ws_size;

    char*  ws  = (char*)d_ws;
    size_t off = 0;
    auto carve = [&](size_t bytes) -> char* {
        char* p = ws + off;
        off += (bytes + 255) & ~(size_t)255;
        return p;
    };
    unsigned short* Wb     = (unsigned short*)carve((size_t)V_DB * D_DIM * 2);
    unsigned short* Cb     = (unsigned short*)carve((size_t)P_C  * D_DIM * 2);
    unsigned short* Xb     = (unsigned short*)carve((size_t)B_Q  * D_DIM * 2);
    float*          qc     = (float*)carve((size_t)B_Q * P_C * 4);
    unsigned*       bits   = (unsigned*)carve((size_t)B_Q * BITW * 4);
    int*            ccount = (int*)carve((size_t)B_Q * 4);
    float*          candS  = (float*)carve((size_t)B_Q * GCAP * 4);
    int*            candI  = (int*)carve((size_t)B_Q * GCAP * 4);

    // k0: precision conversion + candidate-counter reset (every call)
    cvt_f32_bf16<<<2048, 256, 0, stream>>>(W, Wb, V_DB * D_DIM);
    cvt_f32_bf16<<<1024, 256, 0, stream>>>(C, Cb, P_C * D_DIM);
    cvt_f32_bf16<<<256,  256, 0, stream>>>(x, Xb, B_Q * D_DIM);
    zero_counts<<<(B_Q + 255) / 256, 256, 0, stream>>>(ccount, B_Q);

    // k1: coarse quantizer scores
    coarse_gemm<<<dim3(P_C / 64, B_Q / 16), 128, 0, stream>>>(
        (const __bf16*)Xb, (const __bf16*)Cb, qc);

    // k2: probe selection -> bitset
    topk_probe<<<B_Q, 256, 0, stream>>>(qc, bits);

    // k3: masked scan of inverted lists, candidates to global lists
    scan_kernel<<<dim3(B_Q / QT_QUERIES, NSPLIT), 128, 0, stream>>>(
        (const __bf16*)Xb, (const __bf16*)Wb, bits, assign, ccount, candS, candI);

    // k4: exact top-32 -> I (int32) then D (float), concatenated in order
    int*   outI = (int*)d_out;
    float* outD = (float*)d_out + (size_t)B_Q * TOPK;
    final_topk<<<B_Q, 256, 0, stream>>>(ccount, candS, candI, outI, outD);
}
